// Head_1125281431783
// MI455X (gfx1250) — matrix operand order repaired and checked by reading, not compiled, not run on hardware
//
#include <hip/hip_runtime.h>
#include <stdint.h>

typedef _Float16 f16;
typedef __attribute__((ext_vector_type(16))) _Float16 v16h;
typedef __attribute__((ext_vector_type(8)))  _Float16 v8h;
typedef __attribute__((ext_vector_type(8)))  float    v8f;
typedef __attribute__((ext_vector_type(4)))  int      v4i;

#define B_   8
#define T_   1024
#define C_   1024
#define HS_  64
#define NPOS 1024
#define DROWS (2*NPOS-1)   // 2047

#if defined(__has_builtin)
#  if __has_builtin(__builtin_amdgcn_global_load_async_to_lds_b128) && \
      __has_builtin(__builtin_amdgcn_s_wait_asynccnt)
#    define USE_ASYNC_LDS 1
#  endif
#endif
#ifndef USE_ASYNC_LDS
#  define USE_ASYNC_LDS 0
#endif

// Build a 16-bit A-fragment (16x32, K=32): halfs 0..7 from base0, 8..15 from base1.
static __device__ __forceinline__ v16h load_a_frag(const f16* base0, const f16* base1) {
  v8h lo = *(const v8h*)base0;
  v8h hi = *(const v8h*)base1;
  v16h r;
#pragma unroll
  for (int i = 0; i < 8; ++i) { r[i] = lo[i]; r[i + 8] = hi[i]; }
  return r;
}

static __device__ __forceinline__ float hmax16(float v) {
#pragma unroll
  for (int m = 1; m <= 8; m <<= 1) v = fmaxf(v, __shfl_xor(v, m, 32));
  return v;
}
static __device__ __forceinline__ float hsum16(float v) {
#pragma unroll
  for (int m = 1; m <= 8; m <<= 1) v += __shfl_xor(v, m, 32);
  return v;
}

// ---------------- Kernel 0: convert + repack ----------------
__global__ void prepack_kernel(const float* __restrict__ x,
                               const float* __restrict__ Wq,
                               const float* __restrict__ Wk,
                               const float* __restrict__ Wv,
                               const float* __restrict__ demb,
                               f16* __restrict__ xh,
                               f16* __restrict__ Wpack,
                               f16* __restrict__ Eh) {
  const long nx = (long)B_ * T_ * C_;
  const long i0 = (long)blockIdx.x * blockDim.x + threadIdx.x;
  const long stride = (long)gridDim.x * blockDim.x;
  for (long i = i0; i < nx; i += stride) xh[i] = (f16)x[i];
  for (long i = i0; i < (long)DROWS * HS_; i += stride) Eh[i] = (f16)demb[i];
  // W repacked into WMMA B-fragment order:
  // [mat(3)][kchunk(32)][coltile(4)][lane(32)][i(16)]
  const float* Ws[3] = {Wq, Wk, Wv};
  for (long i = i0; i < 3L * 32 * 4 * 32 * 16; i += stride) {
    int ii   = (int)(i & 15);
    int lane = (int)((i >> 4) & 31);
    int t    = (int)((i >> 9) & 3);
    int kc   = (int)((i >> 11) & 31);
    int mat  = (int)(i >> 16);
    int nn   = lane & 15;
    int kr   = (lane < 16) ? 0 : 16;
    Wpack[i] = (f16)Ws[mat][(kc * 32 + kr + ii) * HS_ + t * 16 + nn];
  }
}

// ---------------- Kernel 1: Q/K/V projection (double-buffered WMMA GEMM) ----------------
__global__ void __launch_bounds__(128)
qkv_kernel(const f16* __restrict__ xh, const f16* __restrict__ Wpack,
           f16* __restrict__ Qh, f16* __restrict__ Kh, f16* __restrict__ Vh) {
  const int lane  = threadIdx.x & 31;
  const int w     = threadIdx.x >> 5;
  const int tile  = blockIdx.x * 4 + w;        // 0..511
  const long row0 = (long)tile * 16;
  const int n     = lane & 15;
  const int hi    = (lane >> 4) & 1;
  const int kbase = hi ? 16 : 0;

  v8f acc[12];
#pragma unroll
  for (int i = 0; i < 12; ++i) acc[i] = (v8f){};

  auto loadA = [&](int kc) {
    const f16* xr = xh + (row0 + n) * C_ + kc * 32;
    return load_a_frag(xr + kbase, xr + kbase + 8);
  };

  v16h a = loadA(0);
  v16h bf[12];
#pragma unroll
  for (int i = 0; i < 12; ++i) {
    const int mat = i >> 2, t = i & 3;
    bf[i] = *(const v16h*)(Wpack + ((((long)(mat * 32 + 0) * 4 + t) * 32) + lane) * 16);
  }

  for (int kc = 0; kc < C_ / 32; ++kc) {
    const int kn = (kc + 1 < C_ / 32) ? kc + 1 : kc;
    // prefetch next step's fragments while this step's WMMAs run
    v16h an = loadA(kn);
    v16h bfn[12];
#pragma unroll
    for (int i = 0; i < 12; ++i) {
      const int mat = i >> 2, t = i & 3;
      bfn[i] = *(const v16h*)(Wpack + ((((long)(mat * 32 + kn) * 4 + t) * 32) + lane) * 16);
    }
#pragma unroll
    for (int i = 0; i < 12; ++i)
      acc[i] = __builtin_amdgcn_wmma_f32_16x16x32_f16(
          false, a, false, bf[i], (short)0, acc[i], false, false);
    a = an;
#pragma unroll
    for (int i = 0; i < 12; ++i) bf[i] = bfn[i];
  }

  f16* outs[3] = {Qh, Kh, Vh};
#pragma unroll
  for (int mat = 0; mat < 3; ++mat)
#pragma unroll
    for (int t = 0; t < 4; ++t)
#pragma unroll
      for (int r = 0; r < 8; ++r) {
        long row = row0 + r + 8 * hi;
        outs[mat][row * HS_ + t * 16 + n] = (f16)acc[mat * 4 + t][r];
      }
}

// ---------------- Kernel 2: flash attention with relative-position bias ----------------
__global__ void __launch_bounds__(128)
attn_kernel(const f16* __restrict__ Qh, const f16* __restrict__ Kh,
            const f16* __restrict__ Vh, const f16* __restrict__ Eh,
            const float* __restrict__ amask, float* __restrict__ out) {
  __shared__ float pbias[4][16 * 32];   // bias gather buffer, per wave
  __shared__ f16   pprob[4][16 * 32];   // probability tile (A-layout staging)
  __shared__ f16   vtile[4][32 * 64];   // V tile

  const int lane  = threadIdx.x & 31;
  const int w     = threadIdx.x >> 5;
  const int tile  = blockIdx.x * 4 + w;        // 0..511
  const int b     = tile >> 6;
  const int q0    = (tile & 63) * 16;
  const int n     = lane & 15;
  const int hi    = (lane >> 4) & 1;
  const int kbase = hi ? 16 : 0;
  const int krow  = hi ? 16 : 0;
  const long bT   = (long)b * T_;

  float* pb = pbias[w];
  f16*   pp = pprob[w];
  f16*   vt = vtile[w];

  // Q A-fragments, d-chunks 0 and 32 (A-layout row = lane&15)
  const f16* qrow = Qh + (bT + q0 + n) * HS_;
  const v16h qA0 = load_a_frag(qrow + 0  + kbase, qrow + 0  + kbase + 8);
  const v16h qA1 = load_a_frag(qrow + 32 + kbase, qrow + 32 + kbase + 8);

  v8f accO[4];
#pragma unroll
  for (int t = 0; t < 4; ++t) accO[t] = (v8f){};
  float mrow[8], lrow[8];
#pragma unroll
  for (int r = 0; r < 8; ++r) { mrow[r] = -1e30f; lrow[r] = 0.f; }

  for (int k0 = 0; k0 <= q0 + 15; k0 += 32) {
    // ================= issue all memory for this key block up front =================
    // V tile (32 rows x 64) -> LDS
#if USE_ASYNC_LDS
    {
      v4i* gs = (v4i*)(Vh + (bT + k0 + lane) * HS_);   // 64 halfs = 8 x b128
      v4i* ld = (v4i*)(vt + lane * 64);
#pragma unroll
      for (int c8 = 0; c8 < 8; ++c8)
        __builtin_amdgcn_global_load_async_to_lds_b128(gs + c8, ld + c8, 0, 0);
    }
#else
    {
      const f16* vrp = Vh + (bT + k0 + lane) * HS_;
#pragma unroll
      for (int c8 = 0; c8 < 8; ++c8)
        *(v8h*)(vt + lane * 64 + c8 * 8) = *(const v8h*)(vrp + c8 * 8);
    }
#endif
    // K fragments, rel-emb fragments and attention-mask values for both 16-col tiles
    v16h bk[2][2];
    v16h be[2][2][2];
    float amadd[2];
#pragma unroll
    for (int t01 = 0; t01 < 2; ++t01) {
      const int kc0 = k0 + t01 * 16;
      const f16* krp = Kh + (bT + kc0 + n) * HS_;
      bk[t01][0] = *(const v16h*)(krp + krow);
      bk[t01][1] = *(const v16h*)(krp + 32 + krow);
      const int dbase = q0 - kc0 + (NPOS - 1) - 15;
#pragma unroll
      for (int jt = 0; jt < 2; ++jt) {
        int drow = dbase + jt * 16 + n;
        drow = (drow < 0) ? 0 : (drow > DROWS - 1 ? DROWS - 1 : drow);
        const f16* erp = Eh + (long)drow * HS_;
        be[t01][jt][0] = *(const v16h*)(erp + krow);
        be[t01][jt][1] = *(const v16h*)(erp + 32 + krow);
      }
      amadd[t01] = (1.0f - amask[bT + kc0 + n]) * -3.402823466e38f;
    }
    // prefetch the next key block's K and V rows into cache
    if (k0 + 32 <= q0 + 15) {
      __builtin_prefetch(Kh + (bT + k0 + 32 + n) * HS_, 0, 3);
      __builtin_prefetch(Vh + (bT + k0 + 32 + lane) * HS_, 0, 3);
    }
    // ================= scores =================
    v8f sc[2];
#pragma unroll
    for (int t01 = 0; t01 < 2; ++t01) {
      const int kc0 = k0 + t01 * 16;
      v8f c = {};
      c = __builtin_amdgcn_wmma_f32_16x16x32_f16(false, qA0, false, bk[t01][0], (short)0, c, false, false);
      c = __builtin_amdgcn_wmma_f32_16x16x32_f16(false, qA1, false, bk[t01][1], (short)0, c, false, false);
#pragma unroll
      for (int jt = 0; jt < 2; ++jt) {
        v8f pc = {};
        pc = __builtin_amdgcn_wmma_f32_16x16x32_f16(false, qA0, false, be[t01][jt][0], (short)0, pc, false, false);
        pc = __builtin_amdgcn_wmma_f32_16x16x32_f16(false, qA1, false, be[t01][jt][1], (short)0, pc, false, false);
#pragma unroll
        for (int r = 0; r < 8; ++r) pb[(r + 8 * hi) * 32 + jt * 16 + n] = pc[r];
      }
#pragma unroll
      for (int r = 0; r < 8; ++r) {
        const int M = r + 8 * hi;
        float s = c[r] * 0.125f + pb[M * 32 + (15 + M - n)] + amadd[t01];
        if (kc0 + n > q0 + M) s = -1e30f;   // causal
        sc[t01][r] = s;
      }
    }
    // ================= online softmax =================
    float alpha[8];
#pragma unroll
    for (int r = 0; r < 8; ++r) {
      float bm = hmax16(fmaxf(sc[0][r], sc[1][r]));
      float mn = fmaxf(mrow[r], bm);
      float a  = __expf(mrow[r] - mn);
      float p0 = __expf(sc[0][r] - mn);
      float p1 = __expf(sc[1][r] - mn);
      lrow[r] = lrow[r] * a + hsum16(p0 + p1);
      mrow[r] = mn;
      alpha[r] = a;
      const int M = r + 8 * hi;
      pp[M * 32 + n]      = (f16)p0;
      pp[M * 32 + 16 + n] = (f16)p1;
    }
#pragma unroll
    for (int t = 0; t < 4; ++t)
#pragma unroll
      for (int r = 0; r < 8; ++r) accO[t][r] *= alpha[r];
    // ================= O += P(16x32) * V(32x64) =================
#if USE_ASYNC_LDS
    __builtin_amdgcn_s_wait_asynccnt(0);
#endif
    {
      v16h pA = load_a_frag(pp + n * 32 + kbase, pp + n * 32 + kbase + 8);
#pragma unroll
      for (int t = 0; t < 4; ++t) {
        v16h vB;
#pragma unroll
        for (int i = 0; i < 16; ++i) vB[i] = vt[(krow + i) * 64 + t * 16 + n];
        accO[t] = __builtin_amdgcn_wmma_f32_16x16x32_f16(
            false, pA, false, vB, (short)0, accO[t], false, false);
      }
    }
  }
  // ================= epilogue =================
#pragma unroll
  for (int r = 0; r < 8; ++r) {
    const long row = q0 + r + 8 * hi;
    const float inv = 1.0f / lrow[r];
#pragma unroll
    for (int t = 0; t < 4; ++t)
      out[(bT + row) * HS_ + t * 16 + n] = accO[t][r] * inv;
  }
}

extern "C" void kernel_launch(void* const* d_in, const int* in_sizes, int n_in,
                              void* d_out, int out_size, void* d_ws, size_t ws_size,
                              hipStream_t stream) {
  (void)in_sizes; (void)n_in; (void)out_size; (void)ws_size;
  const float* x  = (const float*)d_in[0];
  const float* Wq = (const float*)d_in[1];
  const float* Wk = (const float*)d_in[2];
  const float* Wv = (const float*)d_in[3];
  const float* de = (const float*)d_in[4];
  const float* am = (const float*)d_in[5];
  float* out = (float*)d_out;

  char* ws = (char*)d_ws;
  size_t off = 0;
  f16* xh = (f16*)(ws + off); off += (size_t)B_ * T_ * C_ * 2;     // 16 MB
  f16* Qh = (f16*)(ws + off); off += (size_t)B_ * T_ * HS_ * 2;    // 1 MB
  f16* Kh = (f16*)(ws + off); off += (size_t)B_ * T_ * HS_ * 2;    // 1 MB
  f16* Vh = (f16*)(ws + off); off += (size_t)B_ * T_ * HS_ * 2;    // 1 MB
  f16* Eh = (f16*)(ws + off); off += ((size_t)DROWS * HS_ * 2 + 255) & ~(size_t)255;
  f16* Wp = (f16*)(ws + off); off += (size_t)3 * 32 * 4 * 32 * 16 * 2;

  prepack_kernel<<<1024, 256, 0, stream>>>(x, Wq, Wk, Wv, de, xh, Wp, Eh);
  qkv_kernel<<<128, 128, 0, stream>>>(xh, Wp, Qh, Kh, Vh);
  attn_kernel<<<128, 128, 0, stream>>>(Qh, Kh, Vh, Eh, am, out);
}
